// Block_90357521973719
// MI455X (gfx1250) — compile-verified, matrix operand order checked by reading, not run on hardware
//
#include <hip/hip_runtime.h>

// ---------------------------------------------------------------------------
// Types / helpers
// ---------------------------------------------------------------------------
typedef __bf16 bf16;
typedef __attribute__((ext_vector_type(8)))  __bf16 bf16x8;
typedef __attribute__((ext_vector_type(16))) __bf16 bf16x16;
typedef __attribute__((ext_vector_type(8)))  float  f32x8;

#define EMBED 1024
#define TSEQ  2048
#define BATCH 2
#define NHEAD 16
#define HSZ   64
#define MROWS (BATCH * TSEQ)   /* 4096 token rows */
#define FFDIM (4 * EMBED)      /* 4096 */

__device__ __forceinline__ bf16 to_bf16(float f) {
  unsigned u = __builtin_bit_cast(unsigned, f);
  unsigned r = u + 0x7FFFu + ((u >> 16) & 1u);   // round-to-nearest-even
  unsigned short h = (unsigned short)(r >> 16);
  return __builtin_bit_cast(bf16, h);
}

__device__ __forceinline__ bf16x16 cat8(bf16x8 lo, bf16x8 hi) {
  bf16x16 r;
#pragma unroll
  for (int i = 0; i < 8; ++i) { r[i] = lo[i]; r[i + 8] = hi[i]; }
  return r;
}

__device__ __forceinline__ f32x8 zero8() {
  f32x8 z;
#pragma unroll
  for (int i = 0; i < 8; ++i) z[i] = 0.0f;
  return z;
}

__device__ __forceinline__ f32x8 wmma_bf16(bf16x16 a, bf16x16 b, f32x8 c) {
  // D = A(16x32) * B(32x16) + C, fp32 accumulate
  return __builtin_amdgcn_wmma_f32_16x16x32_bf16(false, a, false, b,
                                                 (short)0, c, false, false);
}

// --- CDNA5 async global->LDS copy (ASYNCcnt path) --------------------------
// Each active lane copies 16 bytes: GLOBAL[gaddr(lane)] -> LDS[laddr(lane)].
// LDS byte offset = low 32 bits of the generic LDS pointer (flat aperture).
__device__ __forceinline__ void async_g2l_b128(void* lds_dst, const void* gsrc) {
  unsigned l = (unsigned)(unsigned long long)(uintptr_t)lds_dst;
  unsigned long long g = (unsigned long long)(uintptr_t)gsrc;
  asm volatile("global_load_async_to_lds_b128 %0, %1, off"
               :: "v"(l), "v"(g) : "memory");
}
__device__ __forceinline__ void wait_async0() {
  asm volatile("s_wait_asynccnt 0x0" ::: "memory");
}

// ---------------------------------------------------------------------------
// fp32 -> bf16 conversion with transpose: in[K][N] f32 -> out[N][K] bf16.
// Weights are pre-transposed once so GEMM B-tiles stage with contiguous B128
// copies (no scalar ds_store_b16 transpose in the GEMM hot loop).
// ---------------------------------------------------------------------------
__global__ __launch_bounds__(256) void cvtT_kernel(const float* __restrict__ in,
                                                   bf16* __restrict__ out,
                                                   int K, int N) {
  __shared__ bf16 tile[32][33];
  const int kb = blockIdx.x * 32;
  const int nb = blockIdx.y * 32;
  const int tx = threadIdx.x & 31;
  const int ty = threadIdx.x >> 5;   // 0..7
#pragma unroll
  for (int i = 0; i < 4; ++i) {
    int k = ty + i * 8;
    tile[k][tx] = to_bf16(in[(size_t)(kb + k) * N + nb + tx]);
  }
  __syncthreads();
#pragma unroll
  for (int i = 0; i < 4; ++i) {
    int n = ty + i * 8;
    out[(size_t)(nb + n) * K + kb + tx] = tile[tx][n];
  }
}

// ---------------------------------------------------------------------------
// LayerNorm (one row per 256-thread workgroup) fused with bf16 cast
// ---------------------------------------------------------------------------
__global__ __launch_bounds__(256) void ln_kernel(const float* __restrict__ x,
                                                 const float* __restrict__ g,
                                                 const float* __restrict__ bta,
                                                 bf16* __restrict__ out) {
  __shared__ float red[256];
  const int row = blockIdx.x;
  const int tid = threadIdx.x;
  const float4 xv = *(const float4*)&x[(size_t)row * EMBED + tid * 4];

  red[tid] = xv.x + xv.y + xv.z + xv.w;
  __syncthreads();
  for (int st = 128; st > 0; st >>= 1) {
    if (tid < st) red[tid] += red[tid + st];
    __syncthreads();
  }
  const float mu = red[0] * (1.0f / EMBED);
  __syncthreads();

  float d0 = xv.x - mu, d1 = xv.y - mu, d2 = xv.z - mu, d3 = xv.w - mu;
  red[tid] = d0 * d0 + d1 * d1 + d2 * d2 + d3 * d3;
  __syncthreads();
  for (int st = 128; st > 0; st >>= 1) {
    if (tid < st) red[tid] += red[tid + st];
    __syncthreads();
  }
  const float var = red[0] * (1.0f / EMBED);
  const float rs  = rsqrtf(var + 1e-5f);

  const int c0 = tid * 4;
  bf16* op = out + (size_t)row * EMBED + c0;
  op[0] = to_bf16(d0 * rs * g[c0 + 0] + bta[c0 + 0]);
  op[1] = to_bf16(d1 * rs * g[c0 + 1] + bta[c0 + 1]);
  op[2] = to_bf16(d2 * rs * g[c0 + 2] + bta[c0 + 2]);
  op[3] = to_bf16(d3 * rs * g[c0 + 3] + bta[c0 + 3]);
}

// ---------------------------------------------------------------------------
// WMMA GEMM: out[M,N] = epi( A[M,K](bf16) @ Wt[N,K](bf16)^T )
// Block tile 128x128, 8 waves (2 x 4), 64x32 per wave, BK = 32.
// Double-buffered LDS tiles filled via async global->LDS copies: the k+1 slab
// streams in (ASYNCcnt) while WMMAs consume slab k.
// ---------------------------------------------------------------------------
#define GEMM_PAD 40   /* LDS row stride in bf16 elems: 80 B, 16B-aligned */

template <bool BIAS, bool RELU, bool RESID, bool BF16OUT>
__global__ __launch_bounds__(256) void gemm_kernel(
    const bf16* __restrict__ A, const bf16* __restrict__ Wt,
    const float* __restrict__ bias, const float* __restrict__ resid,
    void* __restrict__ outp, int M, int N, int K) {
  __shared__ bf16 As[2][128][GEMM_PAD];   // [m][k]
  __shared__ bf16 Bs[2][128][GEMM_PAD];   // [n][k]

  const int tid  = threadIdx.x;
  const int lane = tid & 31;
  const int wave = tid >> 5;
  const int wm   = wave & 1;   // 0..1 : 64-row strip
  const int wn   = wave >> 1;  // 0..3 : 32-col strip
  const int m0   = blockIdx.x * 128;
  const int n0   = blockIdx.y * 128;
  const int lr   = lane & 15;
  const int koff = (lane < 16) ? 0 : 8;

  // Per-thread staging coordinates (two 16B chunks per tile per thread)
  const int r0 = tid >> 2, c0 = (tid & 3) * 8;           // chunk 0
  const int r1 = (tid + 256) >> 2, c1 = ((tid + 256) & 3) * 8;  // chunk 1

  f32x8 acc[4][2];
#pragma unroll
  for (int mi = 0; mi < 4; ++mi)
#pragma unroll
    for (int ni = 0; ni < 2; ++ni) acc[mi][ni] = zero8();

  const int nk = K >> 5;

  // Prefetch slab 0 into buffer 0
  async_g2l_b128(&As[0][r0][c0], &A[(size_t)(m0 + r0) * K + c0]);
  async_g2l_b128(&As[0][r1][c1], &A[(size_t)(m0 + r1) * K + c1]);
  async_g2l_b128(&Bs[0][r0][c0], &Wt[(size_t)(n0 + r0) * K + c0]);
  async_g2l_b128(&Bs[0][r1][c1], &Wt[(size_t)(n0 + r1) * K + c1]);

  for (int it = 0; it < nk; ++it) {
    const int buf = it & 1;
    wait_async0();
    __syncthreads();
    if (it + 1 < nk) {
      const int nb = buf ^ 1;
      const size_t k0 = (size_t)(it + 1) << 5;
      async_g2l_b128(&As[nb][r0][c0], &A[(size_t)(m0 + r0) * K + k0 + c0]);
      async_g2l_b128(&As[nb][r1][c1], &A[(size_t)(m0 + r1) * K + k0 + c1]);
      async_g2l_b128(&Bs[nb][r0][c0], &Wt[(size_t)(n0 + r0) * K + k0 + c0]);
      async_g2l_b128(&Bs[nb][r1][c1], &Wt[(size_t)(n0 + r1) * K + k0 + c1]);
    }

    bf16x16 af[4];
#pragma unroll
    for (int mi = 0; mi < 4; ++mi) {
      int row = wm * 64 + mi * 16 + lr;
      af[mi] = cat8(*(const bf16x8*)&As[buf][row][koff],
                    *(const bf16x8*)&As[buf][row][koff + 16]);
    }
#pragma unroll
    for (int ni = 0; ni < 2; ++ni) {
      int rb = wn * 32 + ni * 16 + lr;
      bf16x16 bfrag = cat8(*(const bf16x8*)&Bs[buf][rb][koff],
                           *(const bf16x8*)&Bs[buf][rb][koff + 16]);
#pragma unroll
      for (int mi = 0; mi < 4; ++mi)
        acc[mi][ni] = wmma_bf16(af[mi], bfrag, acc[mi][ni]);
    }
    // No trailing barrier needed: next iteration's wait+barrier precedes the
    // async issue that would overwrite this buffer, and all LDS reads here are
    // consumed (DScnt drained) before this wave reaches that barrier.
  }

  const int rhalf = (lane < 16) ? 0 : 8;
#pragma unroll
  for (int mi = 0; mi < 4; ++mi)
#pragma unroll
    for (int ni = 0; ni < 2; ++ni)
#pragma unroll
      for (int r = 0; r < 8; ++r) {
        int gr = m0 + wm * 64 + mi * 16 + r + rhalf;
        int gc = n0 + wn * 32 + ni * 16 + lr;
        float v = acc[mi][ni][r];
        if (BIAS)  v += bias[gc];
        if (RELU)  v = fmaxf(v, 0.0f);
        if (RESID) v += resid[(size_t)gr * N + gc];
        if (BF16OUT) ((bf16*)outp)[(size_t)gr * N + gc] = to_bf16(v);
        else         ((float*)outp)[(size_t)gr * N + gc] = v;
      }
}

// ---------------------------------------------------------------------------
// Flash attention: one workgroup per (b, h, 64-row q block); 4 waves x 16 q
// rows. K tiles double-buffered in LDS via async copies; V prefetched into
// registers one block ahead and committed transposed to LDS. Online softmax
// in fp32; all matmuls on the WMMA pipe.
// ---------------------------------------------------------------------------
#define ATT_PAD 80   /* LDS row stride: 160 B, 16B-aligned */

__global__ __launch_bounds__(128) void attn_kernel(
    const bf16* __restrict__ Q, const bf16* __restrict__ Km,
    const bf16* __restrict__ Vm, bf16* __restrict__ O) {
  __shared__ bf16 Ks[2][64][ATT_PAD];     // K tile, row-major [k-token][d]
  __shared__ bf16 Vt[2][64][ATT_PAD];     // V^T tile: [d][k-token]
  __shared__ bf16 Pt[4][16][ATT_PAD];     // per-wave P tile: [qrow][k-token]

  const int tid   = threadIdx.x;
  const int lane  = tid & 31;
  const int w     = tid >> 5;
  const int lr    = lane & 15;
  const int koff  = (lane < 16) ? 0 : 8;
  const int rhalf = (lane < 16) ? 0 : 8;
  const int qblk  = blockIdx.x;
  const int h     = blockIdx.y;
  const int b     = blockIdx.z;
  const size_t rowbase = (size_t)b * TSEQ + (size_t)qblk * 64;
  const size_t bT = (size_t)b * TSEQ;

  // Q fragments (rows w*16 .. w*16+15, head-dim 0..63 -> 2 K-windows)
  const bf16* qp = Q + (rowbase + w * 16 + lr) * EMBED + h * HSZ;
  const bf16x16 qf0 = cat8(*(const bf16x8*)(qp + koff),
                           *(const bf16x8*)(qp + koff + 16));
  const bf16x16 qf1 = cat8(*(const bf16x8*)(qp + 32 + koff),
                           *(const bf16x8*)(qp + 32 + koff + 16));

  f32x8 Oacc[4];
#pragma unroll
  for (int dj = 0; dj < 4; ++dj) Oacc[dj] = zero8();
  float mrow[8], lrow[8];
#pragma unroll
  for (int r = 0; r < 8; ++r) { mrow[r] = -1e30f; lrow[r] = 0.0f; }

  // Per-thread staging coordinates: 4 chunks of 8 bf16 over a 64x64 tile
  int stok[4], sdc[4];
#pragma unroll
  for (int i = 0; i < 4; ++i) {
    int c = tid + i * 128;
    stok[i] = c >> 3;
    sdc[i]  = (c & 7) * 8;
  }

  // Prefetch block 0: K tile async -> LDS buf0, V tile -> registers
  bf16x8 vreg[4];
#pragma unroll
  for (int i = 0; i < 4; ++i) {
    async_g2l_b128(&Ks[0][stok[i]][sdc[i]],
                   &Km[(bT + stok[i]) * EMBED + h * HSZ + sdc[i]]);
    vreg[i] = *(const bf16x8*)&Vm[(bT + stok[i]) * EMBED + h * HSZ + sdc[i]];
  }

  for (int j = 0; j <= qblk; ++j) {
    const int buf = j & 1;

    // Commit prefetched V registers to Vt[buf] (transposed)
#pragma unroll
    for (int i = 0; i < 4; ++i)
#pragma unroll
      for (int e = 0; e < 8; ++e) Vt[buf][sdc[i] + e][stok[i]] = vreg[i][e];

    wait_async0();
    __syncthreads();

    if (j < qblk) {
      const int nb = buf ^ 1;
      const size_t tbase = bT + (size_t)(j + 1) * 64;
#pragma unroll
      for (int i = 0; i < 4; ++i) {
        async_g2l_b128(&Ks[nb][stok[i]][sdc[i]],
                       &Km[(tbase + stok[i]) * EMBED + h * HSZ + sdc[i]]);
        vreg[i] = *(const bf16x8*)&Vm[(tbase + stok[i]) * EMBED + h * HSZ + sdc[i]];
      }
    }

    // S = Q @ K^T  (16 x 64 per wave) from LDS
    f32x8 S[4];
#pragma unroll
    for (int nj = 0; nj < 4; ++nj) {
      const int tr = nj * 16 + lr;
      bf16x16 kf0 = cat8(*(const bf16x8*)&Ks[buf][tr][koff],
                         *(const bf16x8*)&Ks[buf][tr][koff + 16]);
      bf16x16 kf1 = cat8(*(const bf16x8*)&Ks[buf][tr][32 + koff],
                         *(const bf16x8*)&Ks[buf][tr][32 + koff + 16]);
      S[nj] = wmma_bf16(qf0, kf0, zero8());
      S[nj] = wmma_bf16(qf1, kf1, S[nj]);
    }

    // Online softmax; C/D element r lives at row r+rhalf, col lr.
#pragma unroll
    for (int r = 0; r < 8; ++r) {
      const int qrow = qblk * 64 + w * 16 + r + rhalf;
      float s[4];
      float mx = -1e30f;
#pragma unroll
      for (int nj = 0; nj < 4; ++nj) {
        int kcol = j * 64 + nj * 16 + lr;
        float sv = S[nj][r] * 0.125f;            // 1/sqrt(64)
        if (kcol > qrow) sv = -1e30f;            // causal mask
        s[nj] = sv;
        mx = fmaxf(mx, sv);
      }
#pragma unroll
      for (int off = 8; off >= 1; off >>= 1)
        mx = fmaxf(mx, __shfl_xor(mx, off, 32)); // stays within 16-lane half
      const float mn   = fmaxf(mrow[r], mx);
      const float corr = __expf(mrow[r] - mn);
      float ps = 0.0f;
#pragma unroll
      for (int nj = 0; nj < 4; ++nj) {
        float p = __expf(s[nj] - mn);
        ps += p;
        Pt[w][r + rhalf][nj * 16 + lr] = to_bf16(p);
      }
#pragma unroll
      for (int off = 8; off >= 1; off >>= 1) ps += __shfl_xor(ps, off, 32);
      lrow[r] = lrow[r] * corr + ps;
      mrow[r] = mn;
#pragma unroll
      for (int dj = 0; dj < 4; ++dj) Oacc[dj][r] *= corr;
    }

    // O += P @ V  (A = P from LDS in A-layout, B = V^T rows contiguous)
#pragma unroll
    for (int kh = 0; kh < 2; ++kh) {
      bf16x16 pf = cat8(*(const bf16x8*)&Pt[w][lr][kh * 32 + koff],
                        *(const bf16x8*)&Pt[w][lr][kh * 32 + koff + 16]);
#pragma unroll
      for (int dj = 0; dj < 4; ++dj) {
        bf16x16 vf = cat8(*(const bf16x8*)&Vt[buf][dj * 16 + lr][kh * 32 + koff],
                          *(const bf16x8*)&Vt[buf][dj * 16 + lr][kh * 32 + koff + 16]);
        Oacc[dj] = wmma_bf16(pf, vf, Oacc[dj]);
      }
    }
    // No trailing barrier: next iteration writes only the opposite buffers,
    // and its wait+barrier precede any overwrite of this one.
  }

  // Epilogue: normalize and store bf16 (concat-head layout)
#pragma unroll
  for (int dj = 0; dj < 4; ++dj)
#pragma unroll
    for (int r = 0; r < 8; ++r) {
      float ov = Oacc[dj][r] / lrow[r];
      size_t gr = rowbase + w * 16 + r + rhalf;
      O[gr * EMBED + h * HSZ + dj * 16 + lr] = to_bf16(ov);
    }
}

// ---------------------------------------------------------------------------
// Host-side orchestration
// ---------------------------------------------------------------------------
extern "C" void kernel_launch(void* const* d_in, const int* in_sizes, int n_in,
                              void* d_out, int out_size, void* d_ws,
                              size_t ws_size, hipStream_t stream) {
  (void)in_sizes; (void)n_in; (void)out_size; (void)ws_size;

  const float* x    = (const float*)d_in[0];
  const float* Wq   = (const float*)d_in[1];
  const float* Wk   = (const float*)d_in[2];
  const float* Wv   = (const float*)d_in[3];
  const float* Wo   = (const float*)d_in[4];
  const float* bo   = (const float*)d_in[5];
  const float* W1   = (const float*)d_in[6];
  const float* b1   = (const float*)d_in[7];
  const float* W2   = (const float*)d_in[8];
  const float* b2   = (const float*)d_in[9];
  const float* ln1g = (const float*)d_in[10];
  const float* ln1b = (const float*)d_in[11];
  const float* ln2g = (const float*)d_in[12];
  const float* ln2b = (const float*)d_in[13];

  char* base = (char*)d_ws;
  size_t off = 0;
  auto take = [&](size_t bytes) -> char* {
    char* p = base + off;
    off += (bytes + 255) & ~(size_t)255;
    return p;
  };

  bf16* wq_t   = (bf16*)take((size_t)EMBED * EMBED * 2);   // [N][K]
  bf16* wk_t   = (bf16*)take((size_t)EMBED * EMBED * 2);
  bf16* wv_t   = (bf16*)take((size_t)EMBED * EMBED * 2);
  bf16* wo_t   = (bf16*)take((size_t)EMBED * EMBED * 2);
  bf16* w1_t   = (bf16*)take((size_t)EMBED * FFDIM * 2);   // [4096][1024]
  bf16* w2_t   = (bf16*)take((size_t)FFDIM * EMBED * 2);   // [1024][4096]
  bf16* h_bf   = (bf16*)take((size_t)MROWS * EMBED * 2);
  bf16* q_bf   = (bf16*)take((size_t)MROWS * EMBED * 2);
  bf16* k_bf   = (bf16*)take((size_t)MROWS * EMBED * 2);
  bf16* v_bf   = (bf16*)take((size_t)MROWS * EMBED * 2);
  bf16* att_bf = (bf16*)take((size_t)MROWS * EMBED * 2);
  float* x2    = (float*)take((size_t)MROWS * EMBED * 4);
  bf16* h2_bf  = (bf16*)take((size_t)MROWS * EMBED * 2);
  bf16* ff1_bf = (bf16*)take((size_t)MROWS * FFDIM * 2);

  // 1) weights -> bf16, transposed to [N][K]
  dim3 gWT(EMBED / 32, EMBED / 32);
  cvtT_kernel<<<gWT, 256, 0, stream>>>(Wq, wq_t, EMBED, EMBED);
  cvtT_kernel<<<gWT, 256, 0, stream>>>(Wk, wk_t, EMBED, EMBED);
  cvtT_kernel<<<gWT, 256, 0, stream>>>(Wv, wv_t, EMBED, EMBED);
  cvtT_kernel<<<gWT, 256, 0, stream>>>(Wo, wo_t, EMBED, EMBED);
  cvtT_kernel<<<dim3(EMBED / 32, FFDIM / 32), 256, 0, stream>>>(W1, w1_t, EMBED, FFDIM);
  cvtT_kernel<<<dim3(FFDIM / 32, EMBED / 32), 256, 0, stream>>>(W2, w2_t, FFDIM, EMBED);

  // 2) h = ln1(x) (bf16)
  ln_kernel<<<MROWS, 256, 0, stream>>>(x, ln1g, ln1b, h_bf);

  // 3) Q, K, V projections
  dim3 gD(MROWS / 128, EMBED / 128);
  gemm_kernel<false, false, false, true><<<gD, 256, 0, stream>>>(
      h_bf, wq_t, nullptr, nullptr, q_bf, MROWS, EMBED, EMBED);
  gemm_kernel<false, false, false, true><<<gD, 256, 0, stream>>>(
      h_bf, wk_t, nullptr, nullptr, k_bf, MROWS, EMBED, EMBED);
  gemm_kernel<false, false, false, true><<<gD, 256, 0, stream>>>(
      h_bf, wv_t, nullptr, nullptr, v_bf, MROWS, EMBED, EMBED);

  // 4) causal flash attention
  attn_kernel<<<dim3(TSEQ / 64, NHEAD, BATCH), 128, 0, stream>>>(
      q_bf, k_bf, v_bf, att_bf);

  // 5) x2 = x + attn @ Wo + bo   (fp32)
  gemm_kernel<true, false, true, false><<<gD, 256, 0, stream>>>(
      att_bf, wo_t, bo, x, x2, MROWS, EMBED, EMBED);

  // 6) h2 = ln2(x2) (bf16)
  ln_kernel<<<MROWS, 256, 0, stream>>>(x2, ln2g, ln2b, h2_bf);

  // 7) ff1 = relu(h2 @ W1 + b1) (bf16)
  gemm_kernel<true, true, false, true><<<dim3(MROWS / 128, FFDIM / 128), 256, 0,
                                         stream>>>(
      h2_bf, w1_t, b1, nullptr, ff1_bf, MROWS, FFDIM, EMBED);

  // 8) out = x2 + ff1 @ W2 + b2 (fp32 -> d_out)
  gemm_kernel<true, false, true, false><<<gD, 256, 0, stream>>>(
      ff1_bf, w2_t, b2, x2, (float*)d_out, MROWS, EMBED, FFDIM);
}
